// GATLayer_35390530519977
// MI455X (gfx1250) — matrix operand order repaired and checked by reading, not compiled, not run on hardware
//
#include <hip/hip_runtime.h>
#include <hip/hip_bf16.h>

typedef __attribute__((ext_vector_type(16))) _Float16 v16h;
typedef __attribute__((ext_vector_type(8)))  _Float16 v8h;
typedef __attribute__((ext_vector_type(8)))  float    v8f;

#define GAT_ALPHA 0.2f
#define GAT_LOG2E 1.44269504088896f

// ---------------------------------------------------------------------------
// Kernel 1: Wh = h @ W (f16 WMMA), stored in WMMA-B-fragment blocked layout:
//   WhB[jb][t][lane][idx]  with jb = 32-row block, t = 16-col tile,
//   lane = (k_half*16 + col), idx = k within half (16 halves = 32B contiguous)
// Also computes s = Wh.a_src and d = Wh.a_dst per row (f32).
// One wave = one 16-row tile. Block = 4 waves.
// ---------------------------------------------------------------------------
__global__ __launch_bounds__(128) void gat_wh_scores(
    const float* __restrict__ h, const float* __restrict__ W,
    const float* __restrict__ a_src, const float* __restrict__ a_dst,
    _Float16* __restrict__ WhB, float* __restrict__ s_out,
    float* __restrict__ d_out_v)
{
  const int lane = threadIdx.x & 31;
  const int wave = threadIdx.x >> 5;
  const int q    = blockIdx.x * 4 + wave;   // 16-row tile id over flat [B*N]
  const int m0   = q << 4;
  const int n16  = lane & 15;
  const int half = lane >> 4;

  v8f acc[8] = {};
  const float* hrow = h + (size_t)(m0 + n16) * 128;

  #pragma unroll
  for (int kk = 0; kk < 4; ++kk) {
    const int kb = kk * 32;
    // A fragment (16x32 f16): lane holds row m0+n16; k = (idx/8)*16 + half*8 + idx%8
    float4 a0 = *(const float4*)(hrow + kb + half * 8);
    float4 a1 = *(const float4*)(hrow + kb + half * 8 + 4);
    float4 a2 = *(const float4*)(hrow + kb + 16 + half * 8);
    float4 a3 = *(const float4*)(hrow + kb + 16 + half * 8 + 4);
    v16h A;
    A[0]=(_Float16)a0.x; A[1]=(_Float16)a0.y; A[2]=(_Float16)a0.z; A[3]=(_Float16)a0.w;
    A[4]=(_Float16)a1.x; A[5]=(_Float16)a1.y; A[6]=(_Float16)a1.z; A[7]=(_Float16)a1.w;
    A[8]=(_Float16)a2.x; A[9]=(_Float16)a2.y; A[10]=(_Float16)a2.z; A[11]=(_Float16)a2.w;
    A[12]=(_Float16)a3.x; A[13]=(_Float16)a3.y; A[14]=(_Float16)a3.z; A[15]=(_Float16)a3.w;
    #pragma unroll
    for (int t = 0; t < 8; ++t) {
      // B fragment (32x16 f16): lane holds col n16, k = (idx/8)*16 + half*8 + idx%8
      const float* wp = W + (size_t)(kb + half * 8) * 128 + t * 16 + n16;
      v16h Bf;
      #pragma unroll
      for (int i = 0; i < 16; ++i) Bf[i] = (_Float16)wp[(size_t)(i < 8 ? i : i + 8) * 128];
      acc[t] = __builtin_amdgcn_wmma_f32_16x16x32_f16(false, A, false, Bf,
                                                      (short)0, acc[t], false, false);
    }
  }

  // --- Store Wh in B-fragment blocked layout -------------------------------
  // C-fragment: lane holds col n16, rows 8*half + r.  For j-block jb = q/2,
  // these rows are B-fragment k = p*16 + (8*half + r), p = q&1, held by
  // B-lane (p*16 + n16) at idx (8*half + r) -> one 16-byte store per tile.
  const int jb = q >> 1, p = q & 1;
  #pragma unroll
  for (int t = 0; t < 8; ++t) {
    v8h hv;
    #pragma unroll
    for (int r = 0; r < 8; ++r) hv[r] = (_Float16)acc[t][r];
    _Float16* dst = WhB + (((size_t)jb * 8 + t) * 512
                           + (size_t)(p * 16 + n16) * 16 + half * 8);
    *(v8h*)dst = hv;
  }

  // --- Per-row scores: s_i = Wh[i,:].a_src, d_i = Wh[i,:].a_dst ------------
  float sp[8] = {}, dp[8] = {};
  #pragma unroll
  for (int t = 0; t < 8; ++t) {
    const float as = a_src[t * 16 + n16];
    const float ad = a_dst[t * 16 + n16];
    #pragma unroll
    for (int r = 0; r < 8; ++r) { sp[r] += acc[t][r] * as; dp[r] += acc[t][r] * ad; }
  }
  #pragma unroll
  for (int mask = 1; mask < 16; mask <<= 1) {
    #pragma unroll
    for (int r = 0; r < 8; ++r) {
      sp[r] += __shfl_xor(sp[r], mask, 32);
      dp[r] += __shfl_xor(dp[r], mask, 32);
    }
  }
  if (n16 == 0) {
    #pragma unroll
    for (int r = 0; r < 8; ++r) {
      s_out[m0 + half * 8 + r]   = sp[r];
      d_out_v[m0 + half * 8 + r] = dp[r];
    }
  }
}

// ---------------------------------------------------------------------------
// Kernel 2: flash-style fused softmax-aggregation.
//   h_prime[i,:] = sum_j exp(leaky(s_i+d_j) - log Z_i) * Wh[j,:]
// Block = 4 waves, 64 consecutive rows, all in one batch.  The Wh jb-blocks
// (8 KB each, already in B-fragment layout) are staged in LDS with a double
// buffer and shared by the 4 waves (4x less L2 traffic).  Global loads for
// jb+1 are issued before the exp/VALU work of jb to hide latency; jb+2 is
// warmed with global_prefetch.
// ---------------------------------------------------------------------------
__global__ __launch_bounds__(128) void gat_attn_aggregate(
    const _Float16* __restrict__ WhB, const float* __restrict__ s_in,
    const float* __restrict__ d_in_v, float* __restrict__ out, int N)
{
  __shared__ __align__(32) _Float16 ldsWh[2][4096];   // 2 x 8 KB double buffer

  const int tid  = threadIdx.x;
  const int lane = tid & 31;
  const int wave = tid >> 5;
  const int tile = blockIdx.x * 4 + wave;       // [0, B*N/16)
  const int tilesPerBatch = N >> 4;
  const int b    = tile / tilesPerBatch;
  const int i0   = (tile % tilesPerBatch) << 4;
  const int n16  = lane & 15;
  const int half = lane >> 4;

  const float* db = d_in_v + (size_t)b * N;
  const float  si = s_in[(size_t)b * N + i0 + n16];

  // Phase 1: Z_i = sum_j exp(leaky(si + d_j)); each lane-half scans half of j
  float Z = 0.f;
  {
    const float4* d4 = (const float4*)db;
    const int base = half * (N >> 3);
    #pragma unroll 4
    for (int jj = 0; jj < (N >> 3); ++jj) {
      float4 dv = d4[base + jj];
      float e;
      e = si + dv.x; e = fmaxf(e, GAT_ALPHA * e); Z += exp2f(e * GAT_LOG2E);
      e = si + dv.y; e = fmaxf(e, GAT_ALPHA * e); Z += exp2f(e * GAT_LOG2E);
      e = si + dv.z; e = fmaxf(e, GAT_ALPHA * e); Z += exp2f(e * GAT_LOG2E);
      e = si + dv.w; e = fmaxf(e, GAT_ALPHA * e); Z += exp2f(e * GAT_LOG2E);
    }
  }
  Z += __shfl_xor(Z, 16, 32);
  const float c2 = __log2f(Z);          // p = exp2(e*log2e - c2) = exp(e)/Z

  // Phase 2: WMMA accumulation over j-blocks of 32, Wh staged in LDS
  v8f acc[8] = {};
  const int njb = N >> 5;
  const float4* gsrc = (const float4*)(WhB + (size_t)b * (size_t)njb * 4096);

  // Prologue: cooperatively load jb=0 block (512 float4 = 8 KB)
  {
    float4* dst = (float4*)ldsWh[0];
    #pragma unroll
    for (int u = 0; u < 4; ++u) dst[tid + u * 128] = gsrc[tid + u * 128];
  }

  for (int jb = 0; jb < njb; ++jb) {
    __syncthreads();                    // buf[cur] ready; buf[next] drained
    const int cur = jb & 1;

    // Issue global loads for jb+1 (latency hidden by the exp work below)
    float4 pre[4];
    const bool haveNext = (jb + 1) < njb;
    if (haveNext) {
      const float4* src = gsrc + (size_t)(jb + 1) * 512;
      #pragma unroll
      for (int u = 0; u < 4; ++u) pre[u] = src[tid + u * 128];
    }
    if ((jb + 2) < njb)
      __builtin_prefetch(gsrc + (size_t)(jb + 2) * 512, 0, 1);

    // Build normalized P-tile A-fragment:
    //   idx 0-15 -> j = j0 + half*16 + idx
    const int j0 = jb << 5;
    float4 g0 = *(const float4*)(db + j0 + half * 16);
    float4 g1 = *(const float4*)(db + j0 + half * 16 + 4);
    float4 g2 = *(const float4*)(db + j0 + half * 16 + 8);
    float4 g3 = *(const float4*)(db + j0 + half * 16 + 12);
    const float ev[16] = {g0.x,g0.y,g0.z,g0.w, g1.x,g1.y,g1.z,g1.w,
                          g2.x,g2.y,g2.z,g2.w, g3.x,g3.y,g3.z,g3.w};
    v16h A;
    #pragma unroll
    for (int i = 0; i < 16; ++i) {
      float e = si + ev[i];
      e = fmaxf(e, GAT_ALPHA * e);                   // LeakyReLU
      A[i] = (_Float16)exp2f(e * GAT_LOG2E - c2);    // normalized softmax weight
    }

    // Park the prefetched block in the other LDS buffer
    if (haveNext) {
      float4* dst = (float4*)ldsWh[cur ^ 1];
      #pragma unroll
      for (int u = 0; u < 4; ++u) dst[tid + u * 128] = pre[u];
    }

    // 8 WMMAs against LDS-resident pre-swizzled B fragments
    const _Float16* bb = ldsWh[cur] + (size_t)lane * 16;
    #pragma unroll
    for (int t = 0; t < 8; ++t) {
      v16h Bf = *(const v16h*)(bb + (size_t)t * 512);
      acc[t] = __builtin_amdgcn_wmma_f32_16x16x32_f16(false, A, false, Bf,
                                                      (short)0, acc[t], false, false);
    }
  }

  // Store h_prime (f32, row-major): lane holds col t*16+n16, rows 8*half+r
  float* ob = out + ((size_t)b * N + i0) * 128;
  #pragma unroll
  for (int t = 0; t < 8; ++t) {
    #pragma unroll
    for (int r = 0; r < 8; ++r)
      ob[(size_t)(half * 8 + r) * 128 + t * 16 + n16] = acc[t][r];
  }
}

// ---------------------------------------------------------------------------
extern "C" void kernel_launch(void* const* d_in, const int* in_sizes, int n_in,
                              void* d_out, int out_size, void* d_ws, size_t ws_size,
                              hipStream_t stream) {
  const float* h     = (const float*)d_in[0];
  const float* W     = (const float*)d_in[1];
  const float* a_src = (const float*)d_in[2];
  const float* a_dst = (const float*)d_in[3];
  float* out = (float*)d_out;

  const int Din = 128, N = 4096;
  const int B = in_sizes[0] / (N * Din);          // 4

  _Float16* WhB = (_Float16*)d_ws;                                 // 4 MB f16
  float* s  = (float*)((char*)d_ws + (size_t)B * N * Din * sizeof(_Float16));
  float* dv = s + (size_t)B * N;

  const int tiles = B * N / 16;                    // 1024 waves, 4 waves/block
  gat_wh_scores<<<tiles / 4, 128, 0, stream>>>(h, W, a_src, a_dst, WhB, s, dv);
  gat_attn_aggregate<<<tiles / 4, 128, 0, stream>>>(WhB, s, dv, out, N);
}
